// SelectiveSSM_27685359190822
// MI455X (gfx1250) — matrix operand order repaired and checked by reading, not compiled, not run on hardware
//
#include <hip/hip_runtime.h>
#include <hip/hip_bf16.h>
#include <math.h>

// ---------------- problem constants ----------------
#define BATCH   4
#define SEQ     4096
#define DMODEL  768
#define DINNER  1536
#define DSTATE  16
#define DTRANK  48
#define DTRANKP 64            // dt_rank padded to K%32==0
#define DCONV   4
#define MROWS   (BATCH*SEQ)   // 16384
#define NXZ     (2*DINNER)    // 3072
#define NBC     (DTRANK+2*DSTATE) // 80

typedef __bf16 bf16;
typedef __attribute__((ext_vector_type(16))) bf16  v16bf;
typedef __attribute__((ext_vector_type(8)))  bf16  v8bf;
typedef __attribute__((ext_vector_type(8)))  float v8f;
typedef __attribute__((ext_vector_type(4)))  unsigned int v4u;
typedef __attribute__((ext_vector_type(8)))  int v8i;
typedef __attribute__((ext_vector_type(4)))  int v4i;

__device__ __forceinline__ float bf2f(bf16 b) {
  unsigned short u = __builtin_bit_cast(unsigned short, b);
  unsigned v = ((unsigned)u) << 16;
  return __builtin_bit_cast(float, v);
}
__device__ __forceinline__ bf16 f2bf(float f) {
  unsigned u = __builtin_bit_cast(unsigned, f);
  unsigned r = u + 0x7FFFu + ((u >> 16) & 1u);   // round-to-nearest-even
  unsigned short h = (unsigned short)(r >> 16);
  return __builtin_bit_cast(bf16, h);
}

__device__ __forceinline__ void wait_async0() {
#if __has_builtin(__builtin_amdgcn_s_wait_asynccnt)
  __builtin_amdgcn_s_wait_asynccnt(0);
#else
  asm volatile("s_wait_asynccnt 0x0" ::: "memory");
#endif
}
__device__ __forceinline__ void wait_tensor0() {
#if __has_builtin(__builtin_amdgcn_s_wait_tensorcnt)
  __builtin_amdgcn_s_wait_tensorcnt((short)0);
#else
  asm volatile("s_wait_tensorcnt 0x0" ::: "memory");
#endif
}

// async copy 16B global -> LDS (no VGPR round trip, ASYNCcnt)
__device__ __forceinline__ void async_g2lds_b128(unsigned lds_byte_off, const void* g) {
  asm volatile("global_load_async_to_lds_b128 %0, %1, off"
               :: "v"(lds_byte_off), "v"(g) : "memory");
}

#if __has_builtin(__builtin_amdgcn_tensor_load_to_lds)
#define USE_TDM 1
#else
#define USE_TDM 0
#endif

#if USE_TDM
// TDM: move a [64 rows x 32 elem] bf16 tile (row stride K elems) into LDS,
// padding +16B after every 64B row -> 80B LDS rows (bank-conflict free).
// rows_avail: valid rows from gsrc (OOB rows read as zero).
__device__ __forceinline__ void tdm_load_tile(unsigned lds_off, const bf16* gsrc,
                                              int rows_avail, int K) {
  unsigned long long ga = (unsigned long long)(size_t)gsrc;
  v4u g0;
  g0[0] = 1u;                                             // count=1 (valid)
  g0[1] = lds_off;                                        // lds_addr
  g0[2] = (unsigned)(ga & 0xFFFFFFFFu);                   // global_addr lo
  g0[3] = (unsigned)((ga >> 32) & 0x01FFFFFFu) | (2u << 30); // addr hi | type=2
  unsigned w0 = (1u << 16)    // data_size = 2B
              | (1u << 20)    // pad_enable
              | (3u << 22)    // pad_interval: 16 DWORDs (=64B row)
              | (3u << 25);   // pad_amount:   4 DWORDs (=16B)
  unsigned w1 = ((unsigned)K & 0xFFFFu) << 16;            // tensor_dim0[15:0]
  unsigned w2 = (((unsigned)K >> 16) & 0xFFFFu)           // tensor_dim0[31:16]
              | (((unsigned)rows_avail & 0xFFFFu) << 16); // tensor_dim1[15:0]
  unsigned w3 = (((unsigned)rows_avail >> 16) & 0xFFFFu)  // tensor_dim1[31:16]
              | (32u << 16);                              // tile_dim0 = 32
  unsigned w4 = 64u;                                      // tile_dim1 = 64
  unsigned w5 = (unsigned)K;                              // dim0_stride lo32
  v8i g1 = { (int)w0, (int)w1, (int)w2, (int)w3, (int)w4, (int)w5, 0, 0 };
  v4i g2 = {0, 0, 0, 0};
  v4i g3 = {0, 0, 0, 0};
#if __clang_major__ >= 23
  v8i g4 = {0, 0, 0, 0, 0, 0, 0, 0};
  __builtin_amdgcn_tensor_load_to_lds(g0, g1, g2, g3, g4, 0);
#else
  __builtin_amdgcn_tensor_load_to_lds(g0, g1, g2, g3, 0);
#endif
}
#endif

// ---------------- elementwise converts ----------------
__global__ void k_cvt(const float* __restrict__ in, bf16* __restrict__ out, int n) {
  int i = blockIdx.x * blockDim.x + threadIdx.x;
  int s = gridDim.x * blockDim.x;
  for (; i < n; i += s) out[i] = f2bf(in[i]);
}

__global__ void k_pad_wdt(const float* __restrict__ wdt, bf16* __restrict__ out) {
  int i = blockIdx.x * blockDim.x + threadIdx.x;
  if (i >= DINNER * DTRANKP) return;
  int r = i >> 6, c = i & 63;
  out[i] = (c < DTRANK) ? f2bf(wdt[r * DTRANK + c]) : f2bf(0.f);
}

__global__ void k_split_dr(const float* __restrict__ dbc, bf16* __restrict__ dr) {
  int i = blockIdx.x * blockDim.x + threadIdx.x;
  if (i >= MROWS * DTRANKP) return;
  int r = i >> 6, c = i & 63;
  dr[i] = (c < DTRANK) ? f2bf(dbc[(size_t)r * NBC + c]) : f2bf(0.f);
}

__global__ void k_softplus(float* __restrict__ delta, const float* __restrict__ bdt, int n) {
  int i = blockIdx.x * blockDim.x + threadIdx.x;
  int s = gridDim.x * blockDim.x;
  for (; i < n; i += s) {
    int d = i % DINNER;
    float x = delta[i] + bdt[d];
    delta[i] = (x > 20.f) ? x : log1pf(__expf(x));
  }
}

// ---------------- WMMA GEMM:  C[M,N] = A[M,K](bf16) * W[N,K]^T(bf16) ----------------
// block = 128 threads (4 waves) -> 64x64 C tile; wave w owns rows m0+16w..+15.
// B tile (64n x 32k) staged once per block into LDS via the Tensor Data Mover
// (fallback: per-lane GLOBAL_LOAD_ASYNC_TO_LDS_B128), double-buffered, rows
// padded to 80B for conflict-free ds_load_b128 fragment reads. A fragments and
// B fragments are manually software-pipelined so WMMAs issue back-to-back.
#define BK     32
#define LDSROW 40

template <bool OUT_BF16>
__global__ __launch_bounds__(128)
void k_gemm(const bf16* __restrict__ A, const bf16* __restrict__ W,
            void* __restrict__ Cout, int M, int N, int K) {
  __shared__ bf16 sB[2][64 * LDSROW];          // 2 x 5120 B

  const int tid  = threadIdx.x;
  const int lane = tid & 31;
  const int wave = tid >> 5;
  const int m0 = blockIdx.x * 64 + wave * 16;
  const int n0 = blockIdx.y * 64;
  const int hi   = (lane >> 4) & 1;            // half-wave select
  const int mrow = m0 + (lane & 15);
  const int aoff = hi * 8;                     // A: K chunk {0..7}/{8..15}
  const bf16* arowbase = A + (size_t)mrow * K;

  auto stage = [&](int buf, int kk) {
#if USE_TDM
    if (wave == 0)
      tdm_load_tile((unsigned)(size_t)&sB[buf][0],
                    W + (size_t)n0 * K + kk, N - n0, K);
#else
#pragma unroll
    for (int it = 0; it < 2; ++it) {
      int c   = tid + it * 128;
      int row = c >> 2;
      int cc  = c & 3;
      int nr  = n0 + row;
      if (nr >= N) nr = N - 1;
      async_g2lds_b128((unsigned)(size_t)&sB[buf][row * LDSROW + cc * 8],
                       W + (size_t)nr * K + kk + cc * 8);
    }
#endif
  };
  auto stage_wait = [&]() {
#if USE_TDM
    if (wave == 0) wait_tensor0();
#else
    wait_async0();
#endif
  };
  auto load_a = [&](int kk) -> v16bf {
    const bf16* arow = arowbase + kk + aoff;
    v8bf a0 = *(const v8bf*)(arow);            // K = kk+aoff .. +7
    v8bf a1 = *(const v8bf*)(arow + 16);       // K = kk+aoff+16 .. +23
    return __builtin_shufflevector(a0, a1,
        0,1,2,3,4,5,6,7,8,9,10,11,12,13,14,15);
  };

  v8f acc[4] = {};
  stage(0, 0);
  v16bf af = load_a(0);                        // pipelined A fragment

  for (int kk = 0; kk < K; kk += BK) {
    const int cur = (kk >> 5) & 1;
    stage_wait();                              // staged tile landed in LDS
    __syncthreads();                           // visible to all waves
    if (kk + BK < K) stage(cur ^ 1, kk + BK);  // overlap next stage with math

    // gather all 4 B fragments first (8 ds_load_b128, one wait), then 4 WMMAs
    v16bf bfr[4];
#pragma unroll
    for (int j = 0; j < 4; ++j) {
      const bf16* bp = &sB[cur][(j * 16 + (lane & 15)) * LDSROW + hi * 8];
      v8bf b0 = *(const v8bf*)(bp);
      v8bf b1 = *(const v8bf*)(bp + 16);
      bfr[j] = __builtin_shufflevector(b0, b1,
          0,1,2,3,4,5,6,7,8,9,10,11,12,13,14,15);
    }
    v16bf afn = af;
    if (kk + BK < K) {
      __builtin_prefetch(arowbase + kk + 2 * BK, 0, 1);
      afn = load_a(kk + BK);                   // next A while WMMAs run
    }
#pragma unroll
    for (int j = 0; j < 4; ++j)
      acc[j] = __builtin_amdgcn_wmma_f32_16x16x32_bf16(
          false, af, false, bfr[j], (short)0, acc[j], false, false);
    af = afn;
    __syncthreads();                           // done reading sB[cur]
  }

#pragma unroll
  for (int j = 0; j < 4; ++j) {
    int n = n0 + j * 16 + (lane & 15);
    if (n >= N) continue;
#pragma unroll
    for (int r = 0; r < 8; ++r) {
      int row = m0 + r + hi * 8;
      if (OUT_BF16) ((bf16*)Cout)[(size_t)row * N + n] = f2bf(acc[j][r]);
      else          ((float*)Cout)[(size_t)row * N + n] = acc[j][r];
    }
  }
}

// ---------------- causal depthwise conv + SiLU ----------------
__global__ void k_conv_silu(const bf16* __restrict__ xz, const float* __restrict__ cw,
                            const float* __restrict__ cb, bf16* __restrict__ u, int n) {
  int i = blockIdx.x * blockDim.x + threadIdx.x;
  int s = gridDim.x * blockDim.x;
  for (; i < n; i += s) {
    int d   = i % DINNER;
    int row = i / DINNER;
    int t   = row & (SEQ - 1);
    float acc = cb[d];
#pragma unroll
    for (int j = 0; j < DCONV; ++j) {
      int tj = t - (DCONV - 1) + j;
      if (tj >= 0)
        acc += bf2f(xz[(size_t)(row - (DCONV - 1) + j) * NXZ + d]) * cw[d * DCONV + j];
    }
    float sv = acc / (1.f + __expf(-acc));
    u[i] = f2bf(sv);
  }
}

// ---------------- selective scan: lane owns (b,d), 16 states in regs --------
// B/C rows staged double-buffered via async global->LDS b128 copies.
#define TCHUNK 128
__global__ __launch_bounds__(64)
void k_scan(const float* __restrict__ delta, const bf16* __restrict__ u,
            const float* __restrict__ dbc, const float* __restrict__ A_log,
            float* __restrict__ y) {
  __shared__ float sBC[2][TCHUNK][2 * DSTATE];   // 2 x 16KB
  const int tid = threadIdx.x;
  const int d = blockIdx.x * 64 + tid;
  const int b = blockIdx.y;
  float A[DSTATE], h[DSTATE];
#pragma unroll
  for (int n = 0; n < DSTATE; ++n) {
    A[n] = -__expf(A_log[d * DSTATE + n]);
    h[n] = 0.f;
  }
  const size_t rowbase = (size_t)b * SEQ;

  auto stage = [&](int buf, int t0) {
#pragma unroll
    for (int it = 0; it < 16; ++it) {
      int c = tid + it * 64;                   // 1024 chunks of 16B
      int row = c >> 3, cc = c & 7;
      const float* g = dbc + (rowbase + t0 + row) * NBC + DTRANK + cc * 4;
      async_g2lds_b128((unsigned)(size_t)&sBC[buf][row][cc * 4], g);
    }
  };

  stage(0, 0);
  for (int t0 = 0; t0 < SEQ; t0 += TCHUNK) {
    const int cur = (t0 >> 7) & 1;
    wait_async0();
    __syncthreads();
    if (t0 + TCHUNK < SEQ) stage(cur ^ 1, t0 + TCHUNK);
    for (int tt = 0; tt < TCHUNK; ++tt) {
      size_t idx = (rowbase + t0 + tt) * DINNER + d;
      float dt = delta[idx];
      float du = dt * bf2f(u[idx]);
      float acc = 0.f;
#pragma unroll
      for (int n = 0; n < DSTATE; ++n) {
        float dA = __expf(dt * A[n]);
        h[n] = dA * h[n] + du * sBC[cur][tt][n];
        acc += h[n] * sBC[cur][tt][DSTATE + n];
      }
      y[idx] = acc;
    }
    __syncthreads();
  }
}

// ---------------- y = (y_scan + u*D) * silu(z), -> bf16 ----------------
__global__ void k_gate(const float* __restrict__ y, const bf16* __restrict__ u,
                       const float* __restrict__ D, const bf16* __restrict__ xz,
                       bf16* __restrict__ out, int n) {
  int i = blockIdx.x * blockDim.x + threadIdx.x;
  int s = gridDim.x * blockDim.x;
  for (; i < n; i += s) {
    int d   = i % DINNER;
    int row = i / DINNER;
    float z = bf2f(xz[(size_t)row * NXZ + DINNER + d]);
    float v = (y[i] + bf2f(u[i]) * D[d]) * (z / (1.f + __expf(-z)));
    out[i] = f2bf(v);
  }
}

// ---------------- host launcher ----------------
extern "C" void kernel_launch(void* const* d_in, const int* in_sizes, int n_in,
                              void* d_out, int out_size, void* d_ws, size_t ws_size,
                              hipStream_t stream) {
  const float* x      = (const float*)d_in[0];
  const float* W_in   = (const float*)d_in[1];
  const float* conv_w = (const float*)d_in[2];
  const float* conv_b = (const float*)d_in[3];
  const float* W_x    = (const float*)d_in[4];
  const float* W_dt   = (const float*)d_in[5];
  const float* b_dt   = (const float*)d_in[6];
  const float* A_log  = (const float*)d_in[7];
  const float* Dvec   = (const float*)d_in[8];
  const float* W_out  = (const float*)d_in[9];
  float* out = (float*)d_out;

  char* p = (char*)d_ws;
  auto alloc = [&](size_t bytes) -> char* {
    char* r = p;
    p += (bytes + 255) & ~(size_t)255;
    return r;
  };
  bf16*  Xbf    = (bf16*)alloc((size_t)MROWS * DMODEL * 2);
  bf16*  Winbf  = (bf16*)alloc((size_t)NXZ * DMODEL * 2);
  bf16*  Wxbf   = (bf16*)alloc((size_t)NBC * DINNER * 2);
  bf16*  Wdtbf  = (bf16*)alloc((size_t)DINNER * DTRANKP * 2);
  bf16*  Woutbf = (bf16*)alloc((size_t)DMODEL * DINNER * 2);
  bf16*  xzbf   = (bf16*)alloc((size_t)MROWS * NXZ * 2);
  bf16*  ubf    = (bf16*)alloc((size_t)MROWS * DINNER * 2);
  float* dbc    = (float*)alloc((size_t)MROWS * NBC * 4);
  bf16*  drbf   = (bf16*)alloc((size_t)MROWS * DTRANKP * 2);
  float* delta  = (float*)alloc((size_t)MROWS * DINNER * 4);
  float* yscan  = (float*)alloc((size_t)MROWS * DINNER * 4);
  bf16*  ybf    = (bf16*)alloc((size_t)MROWS * DINNER * 2);

  const int T = 256;
  auto nb = [](long n, int t) { return (int)((n + t - 1) / t); };

  // 1) converts
  k_cvt<<<nb((long)MROWS * DMODEL, T), T, 0, stream>>>(x, Xbf, MROWS * DMODEL);
  k_cvt<<<nb((long)NXZ * DMODEL, T), T, 0, stream>>>(W_in, Winbf, NXZ * DMODEL);
  k_cvt<<<nb((long)NBC * DINNER, T), T, 0, stream>>>(W_x, Wxbf, NBC * DINNER);
  k_pad_wdt<<<nb((long)DINNER * DTRANKP, T), T, 0, stream>>>(W_dt, Wdtbf);
  k_cvt<<<nb((long)DMODEL * DINNER, T), T, 0, stream>>>(W_out, Woutbf, DMODEL * DINNER);

  // 2) in_proj: xz = x @ W_in^T   [16384 x 3072]
  k_gemm<true><<<dim3(MROWS / 64, (NXZ + 63) / 64), 128, 0, stream>>>(
      Xbf, Winbf, xzbf, MROWS, NXZ, DMODEL);

  // 3) depthwise causal conv + SiLU -> u (bf16)
  k_conv_silu<<<nb((long)MROWS * DINNER, T), T, 0, stream>>>(
      xzbf, conv_w, conv_b, ubf, MROWS * DINNER);

  // 4) x_proj: dBC = u @ W_x^T   [16384 x 80]
  k_gemm<false><<<dim3(MROWS / 64, (NBC + 63) / 64), 128, 0, stream>>>(
      ubf, Wxbf, dbc, MROWS, NBC, DINNER);

  // 5) dt path: pad d_r to K=64, GEMM, softplus(+bias)
  k_split_dr<<<nb((long)MROWS * DTRANKP, T), T, 0, stream>>>(dbc, drbf);
  k_gemm<false><<<dim3(MROWS / 64, (DINNER + 63) / 64), 128, 0, stream>>>(
      drbf, Wdtbf, delta, MROWS, DINNER, DTRANKP);
  k_softplus<<<nb((long)MROWS * DINNER, T), T, 0, stream>>>(delta, b_dt, MROWS * DINNER);

  // 6) selective scan
  k_scan<<<dim3(DINNER / 64, BATCH), 64, 0, stream>>>(delta, ubf, dbc, A_log, yscan);

  // 7) gate: (y + u*D) * silu(z) -> bf16
  k_gate<<<nb((long)MROWS * DINNER, T), T, 0, stream>>>(
      yscan, ubf, Dvec, xzbf, ybf, MROWS * DINNER);

  // 8) out_proj: out = y @ W_out^T   [16384 x 768] f32
  k_gemm<false><<<dim3(MROWS / 64, (DMODEL + 63) / 64), 128, 0, stream>>>(
      ybf, Woutbf, out, MROWS, DMODEL, DINNER);
}
